// MultiHeadAttention_63342177681759
// MI455X (gfx1250) — matrix operand order repaired and checked by reading, not compiled, not run on hardware
//
#include <hip/hip_runtime.h>
#include <hip/hip_bf16.h>

// ---- problem constants (match reference) ----
constexpr int Bc  = 2;
constexpr int Lc  = 2048;
constexpr int Dc  = 1024;
constexpr int Hc  = 16;
constexpr int HDc = 64;          // head dim
constexpr int Mtot = Bc * Lc;    // 4096 rows

typedef __bf16 bf16;
typedef bf16  v8bf  __attribute__((ext_vector_type(8)));
typedef bf16  v16bf __attribute__((ext_vector_type(16)));
typedef float v8f   __attribute__((ext_vector_type(8)));

typedef unsigned int u32x4 __attribute__((ext_vector_type(4)));
typedef int          i32x4 __attribute__((ext_vector_type(4)));
typedef int          i32x8 __attribute__((ext_vector_type(8)));

// Build a 16-element bf16 fragment from two contiguous 8-element chunks.
__device__ inline v16bf make_frag(const bf16* p0, const bf16* p1) {
    v8bf lo = *(const v8bf*)p0;
    v8bf hi = *(const v8bf*)p1;
    v16bf r;
#pragma unroll
    for (int i = 0; i < 8; ++i) { r[i] = lo[i]; r[i + 8] = hi[i]; }
    return r;
}

// A-fragment (16x32, M rows in lanes): lane l16 = row mo+l16,
// K chunks at ko+8*half and ko+16+8*half (ISA 16-bit A layout).
__device__ inline v16bf load_frag_a(const bf16* base, int lda, int mo, int ko,
                                    int l16, int half) {
    const bf16* p0 = base + (size_t)(mo + l16) * lda + ko + 8 * half;
    return make_frag(p0, p0 + 16);
}

// ---------------- TDM: load a [64 rows x 64B] tile (row stride = 2KB) into
// LDS with a 16B pad after every 64B row (LDS row stride becomes 80B).
// D# packing per CDNA5 ISA 8.3-8.6: data_size=4B, tile_dim0=16 units,
// tile_dim1=64, tensor_dim0_stride=512 units, pad_interval=3 (16 DW),
// pad_amount=3 (4 DW).
__device__ inline void tdm_load_wtile(unsigned lds_off,
                                      unsigned long long gaddr) {
    u32x4 g0;
    g0[0] = 1u;                                        // count=1, user mode
    g0[1] = lds_off;                                   // lds_addr
    g0[2] = (unsigned)(gaddr & 0xffffffffull);         // global_addr lo
    g0[3] = (unsigned)(gaddr >> 32) | (2u << 30);      // global_addr hi|type=2
    i32x8 g1;
    g1[0] = (2 << 16) | (1 << 20) | (3 << 22) | (3 << 25);
    g1[1] = (int)((512u & 0xffffu) << 16);             // tensor_dim0 lo16
    g1[2] = 0;                                         // dim0 hi / dim1 lo16
    g1[3] = (int)(0x10u | (16u << 16));                // dim1 hi16 | tile_dim0
    g1[4] = 64;                                        // tile_dim1 | tile_dim2=0
    g1[5] = 512;                                       // dim0_stride lo32
    g1[6] = 0;
    g1[7] = 0;
    i32x4 z4 = {0, 0, 0, 0};
#if defined(__clang_major__) && (__clang_major__ >= 23)
    i32x8 z8 = {0, 0, 0, 0, 0, 0, 0, 0};
    __builtin_amdgcn_tensor_load_to_lds(g0, g1, z4, z4, z8, 0);
#else
    __builtin_amdgcn_tensor_load_to_lds(g0, g1, z4, z4, 0);
#endif
}

// ---------------- cooperative async copy: 64 rows x 128B -> LDS rows of
// 144B (72 bf16) for conflict-free b128 fragment reads. 128 threads,
// 4 x b128 per thread = 8KB, tracked on ASYNCcnt.
__device__ inline void stage_async_64x64(const bf16* gbase, size_t grow_stride,
                                         bf16* lds_base, int tid) {
#pragma unroll
    for (int it = 0; it < 4; ++it) {
        int id = tid + 128 * it;      // 0..511  (16B chunks)
        int r = id >> 3;              // row 0..63
        int c = id & 7;               // 16B chunk in row
        unsigned loff = (unsigned)(uintptr_t)(lds_base + r * 72 + c * 8);
        unsigned long long ga =
            (unsigned long long)(uintptr_t)(gbase + (size_t)r * grow_stride +
                                            c * 8);
        asm volatile("global_load_async_to_lds_b128 %0, %1, off"
                     :
                     : "v"(loff), "v"(ga)
                     : "memory");
    }
}

// ---------------- fp32 -> bf16 cast ----------------
__global__ void cast_f32_bf16_kernel(const float* __restrict__ in,
                                     bf16* __restrict__ out, int n) {
    int i = blockIdx.x * blockDim.x + threadIdx.x;
    if (i < n) out[i] = (bf16)in[i];
}

// ---------------- WMMA GEMM: C = A * W^T + bias ----------------
// W tile staged through LDS by the Tensor Data Mover, double-buffered.
// MODE 0: fp32 out [M x N]                      (final projection)
// MODE 1: RoPE + bf16 out [B,H,L,HD]            (Q / K projection)
// MODE 2: bf16 out transposed [B,H,HD,L]        (V projection)
template <int MODE>
__global__ __launch_bounds__(256) void gemm_bf16_kernel(
    const bf16* __restrict__ A, const bf16* __restrict__ W,
    const float* __restrict__ bias, const float* __restrict__ sinT,
    const float* __restrict__ cosT, float* __restrict__ outF,
    bf16* __restrict__ outB, int M, int N, int K) {
    __shared__ bf16 Bs[2][64 * 40];  // 64 rows x (32 bf16 + 8 pad) = 80B rows

    const int lane = threadIdx.x & 31;
    const int w    = threadIdx.x >> 5;
    const int half = lane >> 4;
    const int l16  = lane & 15;
    const int mo = blockIdx.y * 128 + w * 16;   // wave's 16 rows
    const int no = blockIdx.x * 64;             // block's 64 cols (one head)

    // prologue: TDM stage of first W tile (wave 0 only; EXEC-independent DMA)
    if (w == 0) {
        unsigned long long ga =
            (unsigned long long)(uintptr_t)(W + ((size_t)no * K));
        tdm_load_wtile((unsigned)(uintptr_t)&Bs[0][0], ga);
    }

    v8f acc[4] = {};
    int buf = 0;
    for (int ko = 0; ko < K; ko += 32) {
        if (w == 0) {
            if (ko + 32 < K) {
                unsigned long long ga = (unsigned long long)(uintptr_t)(
                    W + ((size_t)no * K + ko + 32));
                tdm_load_wtile((unsigned)(uintptr_t)&Bs[buf ^ 1][0], ga);
                __builtin_amdgcn_s_wait_tensorcnt(1);
            } else {
                __builtin_amdgcn_s_wait_tensorcnt(0);
            }
        }
        __syncthreads();
        v16bf a = load_frag_a(A, K, mo, ko, l16, half);
        // batch all four B fragments first (independent ds_load clauses),
        // then issue the WMMAs -> no per-WMMA s_wait_dscnt 0x0 serialization
        v16bf bfr[4];
#pragma unroll
        for (int t = 0; t < 4; ++t) {
            const bf16* p0 = &Bs[buf][(t * 16 + l16) * 40 + 8 * half];
            bfr[t] = make_frag(p0, p0 + 16);
        }
#pragma unroll
        for (int t = 0; t < 4; ++t)
            acc[t] = __builtin_amdgcn_wmma_f32_16x16x32_bf16(
                false, a, false, bfr[t], (short)0, acc[t], false, false);
        __syncthreads();
        buf ^= 1;
    }

    // bias
    float vals[4][8];
#pragma unroll
    for (int t = 0; t < 4; ++t) {
        float bv = bias[no + 16 * t + l16];
#pragma unroll
        for (int i = 0; i < 8; ++i) vals[t][i] = acc[t][i] + bv;
    }

    if (MODE == 0) {
#pragma unroll
        for (int i = 0; i < 8; ++i) {
            int m = mo + half * 8 + i;
#pragma unroll
            for (int t = 0; t < 4; ++t)
                outF[(size_t)m * N + no + 16 * t + l16] = vals[t][i];
        }
    } else {
        const int h = no / HDc;   // 64-wide tile == one head
#pragma unroll
        for (int i = 0; i < 8; ++i) {
            int m = mo + half * 8 + i;
            int b = m / Lc, l = m % Lc;
            if (MODE == 1) {
                // RoPE: pairs (j, j+32) live in tiles (t, t+2), same lane.
#pragma unroll
                for (int t = 0; t < 2; ++t) {
                    int j = t * 16 + l16;
                    float s_ = sinT[l * (HDc / 2) + j];
                    float c_ = cosT[l * (HDc / 2) + j];
                    float q1 = vals[t][i], q2 = vals[t + 2][i];
                    vals[t][i]     = q1 * c_ - q2 * s_;
                    vals[t + 2][i] = q2 * c_ + q1 * s_;
                }
                size_t rowbase = (((size_t)b * Hc + h) * Lc + l) * HDc;
#pragma unroll
                for (int t = 0; t < 4; ++t)
                    outB[rowbase + t * 16 + l16] = (bf16)vals[t][i];
            } else {  // MODE == 2 : V transposed [B,H,HD,L]
                size_t base = ((size_t)b * Hc + h) * (size_t)HDc * Lc;
#pragma unroll
                for (int t = 0; t < 4; ++t)
                    outB[base + (size_t)(t * 16 + l16) * Lc + l] =
                        (bf16)vals[t][i];
            }
        }
    }
}

// ---------------- flash attention ----------------
// One block = 4 waves = 64 query rows of one (b,h). K/V 64x64 tiles are
// staged into padded LDS with async global->LDS copies (ASYNCcnt),
// double-buffered; P re-shaped C-layout -> A-fragment via per-wave LDS.
__global__ __launch_bounds__(128) void flash_attn_kernel(
    const bf16* __restrict__ Qb, const bf16* __restrict__ Kb,
    const bf16* __restrict__ Vt, const unsigned char* __restrict__ mask,
    bf16* __restrict__ outB) {
    __shared__ bf16 Ks[2][64 * 72];   // keys x dims, 144B rows
    __shared__ bf16 Vs[2][64 * 72];   // dims x keys, 144B rows
    __shared__ bf16 plds[4][16][72];  // per-wave P tile

    const int tid  = threadIdx.x;
    const int lane = tid & 31;
    const int w    = tid >> 5;
    const int half = lane >> 4;
    const int l16  = lane & 15;

    const int nQB  = Lc / 64;
    int bid  = blockIdx.x;
    int qblk = bid % nQB;
    int h    = (bid / nQB) % Hc;
    int b    = bid / (nQB * Hc);
    const int qo = qblk * 64 + w * 16;  // wave's 16 query rows (within L)

    const size_t headoff = ((size_t)b * Hc + h) * (size_t)Lc * HDc;
    const bf16* Qh = Qb + headoff;           // [L][HD]
    const bf16* Kh = Kb + headoff;           // [L][HD]
    const bf16* Vh = Vt + headoff;           // [HD][L]
    const unsigned char* mrow = mask + (size_t)b * Lc;

    // Q fragments (HD=64 -> 2 K-steps), reused for every KV block.
    v16bf qa[2];
#pragma unroll
    for (int kk = 0; kk < 2; ++kk) {
        const bf16* p0 = Qh + (size_t)(qo + l16) * HDc + kk * 32 + 8 * half;
        qa[kk] = make_frag(p0, p0 + 16);
    }

    // prologue stage of first KV block
    stage_async_64x64(Kh, HDc, &Ks[0][0], tid);
    stage_async_64x64(Vh, Lc, &Vs[0][0], tid);

    v8f o[4] = {};
    float m_i[8], l_i[8];
#pragma unroll
    for (int i = 0; i < 8; ++i) { m_i[i] = -1e30f; l_i[i] = 0.f; }
    const float scale = 0.125f;  // HD^-0.5

    int buf = 0;
    for (int kb = 0; kb < Lc; kb += 64) {
        // stage next KV block, wait for current one (per-wave ASYNCcnt)
        int nkb = kb + 64;
        if (nkb < Lc) {
            stage_async_64x64(Kh + (size_t)nkb * HDc, HDc, &Ks[buf ^ 1][0], tid);
            stage_async_64x64(Vh + nkb, Lc, &Vs[buf ^ 1][0], tid);
            asm volatile("s_wait_asynccnt 0x8" ::: "memory");
        } else {
            asm volatile("s_wait_asynccnt 0x0" ::: "memory");
        }
        __syncthreads();

        // S = Q * K^T (16x64 per wave), K fragments from LDS.
        // Batch the 4 fragment loads per k-step, then the 4 WMMAs.
        v8f s[4] = {};
#pragma unroll
        for (int kk = 0; kk < 2; ++kk) {
            v16bf kf[4];
#pragma unroll
            for (int t = 0; t < 4; ++t) {
                const bf16* p0 =
                    &Ks[buf][(t * 16 + l16) * 72 + kk * 32 + 8 * half];
                kf[t] = make_frag(p0, p0 + 16);
            }
#pragma unroll
            for (int t = 0; t < 4; ++t)
                s[t] = __builtin_amdgcn_wmma_f32_16x16x32_bf16(
                    false, qa[kk], false, kf[t], (short)0, s[t], false, false);
        }
        // scale + mask
        float sv[4][8];
#pragma unroll
        for (int t = 0; t < 4; ++t) {
            bool ok = mrow[kb + t * 16 + l16] != 0;
#pragma unroll
            for (int i = 0; i < 8; ++i)
                sv[t][i] = ok ? s[t][i] * scale : -1e30f;
        }
        // row max (across 16 lanes of the half-wave) + online rescale
        float mnew[8], alpha[8];
#pragma unroll
        for (int i = 0; i < 8; ++i) {
            float v = fmaxf(fmaxf(sv[0][i], sv[1][i]),
                            fmaxf(sv[2][i], sv[3][i]));
#pragma unroll
            for (int off = 1; off < 16; off <<= 1)
                v = fmaxf(v, __shfl_xor(v, off, 32));
            float mn = fmaxf(m_i[i], v);
            mnew[i]  = mn;
            alpha[i] = __expf(m_i[i] - mn);
            m_i[i]   = mn;
        }
        // P = exp(S - m), row sums
        float rs[8];
#pragma unroll
        for (int i = 0; i < 8; ++i) rs[i] = 0.f;
#pragma unroll
        for (int t = 0; t < 4; ++t)
#pragma unroll
            for (int i = 0; i < 8; ++i) {
                float p = __expf(sv[t][i] - mnew[i]);
                sv[t][i] = p;
                rs[i] += p;
            }
#pragma unroll
        for (int i = 0; i < 8; ++i) {
            float v = rs[i];
#pragma unroll
            for (int off = 1; off < 16; off <<= 1)
                v += __shfl_xor(v, off, 32);
            l_i[i] = l_i[i] * alpha[i] + v;
#pragma unroll
            for (int t = 0; t < 4; ++t) o[t][i] *= alpha[i];
        }
        // C-layout -> row-major bf16 P tile in per-wave LDS region
        __syncthreads();
#pragma unroll
        for (int t = 0; t < 4; ++t)
#pragma unroll
            for (int i = 0; i < 8; ++i)
                plds[w][half * 8 + i][t * 16 + l16] = (bf16)sv[t][i];
        __syncthreads();
        // O += P * V (V fragments from LDS; batch loads, then WMMAs)
#pragma unroll
        for (int kk = 0; kk < 2; ++kk) {
            const bf16* p0 = &plds[w][l16][kk * 32 + 8 * half];
            v16bf pa = make_frag(p0, p0 + 16);
            v16bf vf[4];
#pragma unroll
            for (int t = 0; t < 4; ++t) {
                const bf16* v0 =
                    &Vs[buf][(t * 16 + l16) * 72 + kk * 32 + 8 * half];
                vf[t] = make_frag(v0, v0 + 16);
            }
#pragma unroll
            for (int t = 0; t < 4; ++t)
                o[t] = __builtin_amdgcn_wmma_f32_16x16x32_bf16(
                    false, pa, false, vf[t], (short)0, o[t], false, false);
        }
        __syncthreads();  // all waves done with buf before it is re-staged
        buf ^= 1;
    }

    // normalize, store bf16 into [B*L, D] (A-operand of final projection)
#pragma unroll
    for (int i = 0; i < 8; ++i) {
        float inv = l_i[i] > 0.f ? 1.f / l_i[i] : 0.f;
        int l = qo + half * 8 + i;
        size_t rowbase = ((size_t)b * Lc + l) * Dc + h * HDc;
#pragma unroll
        for (int t = 0; t < 4; ++t)
            outB[rowbase + t * 16 + l16] = (bf16)(o[t][i] * inv);
    }
}

extern "C" void kernel_launch(void* const* d_in, const int* in_sizes, int n_in,
                              void* d_out, int out_size, void* d_ws,
                              size_t ws_size, hipStream_t stream) {
    const float* x    = (const float*)d_in[0];
    const float* sinT = (const float*)d_in[1];
    const float* cosT = (const float*)d_in[2];
    const unsigned char* mask = (const unsigned char*)d_in[3];
    const float* Wq = (const float*)d_in[4];
    const float* bq = (const float*)d_in[5];
    const float* Wk = (const float*)d_in[6];
    const float* bk = (const float*)d_in[7];
    const float* Wv = (const float*)d_in[8];
    const float* bv = (const float*)d_in[9];
    const float* Wo = (const float*)d_in[10];
    const float* bo = (const float*)d_in[11];
    float* out = (float*)d_out;

    // workspace carve-up
    char* ws = (char*)d_ws;
    bf16* xb  = (bf16*)ws; ws += (size_t)Mtot * Dc * sizeof(bf16);
    bf16* wqb = (bf16*)ws; ws += (size_t)Dc * Dc * sizeof(bf16);
    bf16* wkb = (bf16*)ws; ws += (size_t)Dc * Dc * sizeof(bf16);
    bf16* wvb = (bf16*)ws; ws += (size_t)Dc * Dc * sizeof(bf16);
    bf16* wob = (bf16*)ws; ws += (size_t)Dc * Dc * sizeof(bf16);
    bf16* Qb  = (bf16*)ws; ws += (size_t)Mtot * Dc * sizeof(bf16);
    bf16* Kb  = (bf16*)ws; ws += (size_t)Mtot * Dc * sizeof(bf16);
    bf16* Vt  = (bf16*)ws; ws += (size_t)Mtot * Dc * sizeof(bf16);
    bf16* ao  = (bf16*)ws; ws += (size_t)Mtot * Dc * sizeof(bf16);

    // 1) bf16 casts
    int nx = Mtot * Dc;
    cast_f32_bf16_kernel<<<(nx + 255) / 256, 256, 0, stream>>>(x, xb, nx);
    int nw = Dc * Dc;
    cast_f32_bf16_kernel<<<(nw + 255) / 256, 256, 0, stream>>>(Wq, wqb, nw);
    cast_f32_bf16_kernel<<<(nw + 255) / 256, 256, 0, stream>>>(Wk, wkb, nw);
    cast_f32_bf16_kernel<<<(nw + 255) / 256, 256, 0, stream>>>(Wv, wvb, nw);
    cast_f32_bf16_kernel<<<(nw + 255) / 256, 256, 0, stream>>>(Wo, wob, nw);

    // 2) Q/K/V projections (RoPE fused for Q,K; V stored transposed)
    dim3 gg(Dc / 64, Mtot / 128);  // (16, 32)
    gemm_bf16_kernel<1><<<gg, 256, 0, stream>>>(xb, wqb, bq, sinT, cosT,
                                                nullptr, Qb, Mtot, Dc, Dc);
    gemm_bf16_kernel<1><<<gg, 256, 0, stream>>>(xb, wkb, bk, sinT, cosT,
                                                nullptr, Kb, Mtot, Dc, Dc);
    gemm_bf16_kernel<2><<<gg, 256, 0, stream>>>(xb, wvb, bv, nullptr, nullptr,
                                                nullptr, Vt, Mtot, Dc, Dc);

    // 3) flash attention
    flash_attn_kernel<<<Bc * Hc * (Lc / 64), 128, 0, stream>>>(Qb, Kb, Vt,
                                                               mask, ao);

    // 4) output projection -> fp32 d_out
    gemm_bf16_kernel<0><<<gg, 256, 0, stream>>>(ao, wob, bo, nullptr, nullptr,
                                                out, nullptr, Mtot, Dc, Dc);
}
